// LSTM_67972152427197
// MI455X (gfx1250) — matrix operand order repaired and checked by reading, not compiled, not run on hardware
//
#include <hip/hip_runtime.h>
#include <hip/hip_bf16.h>

// ---------------------------------------------------------------------------
// LSTM (B=256, T=512, H=512, D=1, C=10) for gfx1250 (MI455X, wave32, WMMA).
//
// v3: register-resident weights. 16 batch groups x 8 H-slices = 128 WGs,
// 4 waves each (one per SIMD32). Each wave owns one 16-column j-tile and
// holds all of its recurrent weights -- 4 gates x 16 k-tiles of
// v_wmma_f32_16x16x32_bf16 B-fragments = 512 VGPRs -- in registers for the
// whole 512-step scan (CDNA5 1024-VGPR / VGPR-MSB feature). LDS holds only
// the 16 KB h tile; steady state per k-tile is 2 ds_load_b128 + 4 WMMAs.
// WGs of a batch group sync via a monotonic device-scope atomic barrier on a
// double-buffered global h array.
// ---------------------------------------------------------------------------

typedef __attribute__((ext_vector_type(16))) __bf16        v16bf;
typedef __attribute__((ext_vector_type(8)))  float         v8f;
typedef __attribute__((ext_vector_type(4)))  unsigned int  u32x4;
typedef __attribute__((ext_vector_type(8)))  unsigned int  u32x8;

struct A32 { u32x4 lo; u32x4 hi; };   // 32 bytes -> one v16bf fragment

constexpr int Bsz   = 256;
constexpr int T     = 512;
constexpr int H     = 512;
constexpr int Cout  = 10;
constexpr int TB    = 16;             // batch rows per group
constexpr int NS    = 8;              // H slices (workgroups) per batch group
constexpr int JT_WG = 4;              // 16-col j-tiles per WG (64 cols)
constexpr int NTHR  = 128;            // 4 waves per WG (one wave per j-tile)

__device__ __forceinline__ unsigned short f2bf(float f) {
    unsigned int u = __builtin_bit_cast(unsigned int, f);
    u += 0x7FFFu + ((u >> 16) & 1u);          // round-to-nearest-even
    return (unsigned short)(u >> 16);
}
__device__ __forceinline__ float bf2f(unsigned short v) {
    return __builtin_bit_cast(float, (unsigned int)v << 16);
}
__device__ __forceinline__ float sigm(float v) {
    return 1.0f / (1.0f + __expf(-v));
}
__device__ __forceinline__ float ftanh(float v) {
    float e = __expf(-2.0f * v);
    return (1.0f - e) / (1.0f + e);
}

// ---------------------------------------------------------------------------
// Zero the double-buffered h exchange array + barrier counters (re-run every
// launch so graph replays are deterministic).
// ---------------------------------------------------------------------------
__global__ void lstm_init(unsigned int* zbase, int nwords) {
    int i = blockIdx.x * blockDim.x + threadIdx.x;
    if (i < nwords) zbase[i] = 0u;
}

// ---------------------------------------------------------------------------
// Pack the four HxH recurrent matrices (f32, row-major W[k][j]) into bf16
// wave32 B-fragments:
//   flat = (((gate*32 + jt)*16 + kt)*32 + lane)*16 + i
//   K = kt*32 + (lane/16)*16 + i ,  N = jt*16 + (lane%16)
// Each WMMA lane reads its 16 bf16 (32 B) contiguously.
// ---------------------------------------------------------------------------
__global__ void lstm_pack_weights(const float* __restrict__ Wg,
                                  const float* __restrict__ Wi,
                                  const float* __restrict__ Wf,
                                  const float* __restrict__ Wo,
                                  unsigned short* __restrict__ pw) {
    int p = blockIdx.x * blockDim.x + threadIdx.x;
    if (p >= 4 * H * H) return;
    int gate = p >> 18;               // H*H = 2^18
    int rem  = p & 262143;
    int jt   = rem >> 13;
    int rem2 = rem & 8191;
    int kt   = rem2 >> 9;
    int rem3 = rem2 & 511;
    int lane = rem3 >> 4;
    int i    = rem3 & 15;
    int K = kt * 32 + ((lane >> 4) << 4) + i;
    int N = jt * 16 + (lane & 15);
    const float* W = (gate == 0) ? Wg : (gate == 1) ? Wi : (gate == 2) ? Wf : Wo;
    pw[p] = f2bf(W[K * H + N]);
}

// ---------------------------------------------------------------------------
// Persistent scan. grid = 16 groups x 8 slices = 128 blocks, 128 threads.
// ---------------------------------------------------------------------------
__global__ __launch_bounds__(NTHR)
__attribute__((amdgpu_waves_per_eu(1)))
void lstm_scan_split(const float* __restrict__ x,
                     const float* __restrict__ Wxg, const float* __restrict__ Wxi,
                     const float* __restrict__ Wxf, const float* __restrict__ Wxo,
                     const float* __restrict__ bg,  const float* __restrict__ bi,
                     const float* __restrict__ bf_, const float* __restrict__ bo,
                     const unsigned short* __restrict__ pw,
                     unsigned short* __restrict__ hb,      // [2][256][512] bf16
                     unsigned int* __restrict__ cnt) {     // [16] barrier ctrs
    __shared__ __align__(16) unsigned short hsh[TB * H];   // 16 KB h tile
    __shared__ float xsh[TB];

    const int tid    = threadIdx.x;
    const int lane   = tid & 31;
    const int w      = tid >> 5;      // wave id == local j-tile
    const int laneHi = lane >> 4;
    const int laneN  = lane & 15;
    const int group  = blockIdx.x >> 3;
    const int s      = blockIdx.x & 7;
    const int b0     = group * TB;
    const int cb     = s * 64 + w * 16;         // this wave's column base
    unsigned int* mycnt = cnt + group;

    // ---- one-time: pull this wave's entire weight slice into VGPRs.
    // 4 gates x 16 k-tiles x 8 VGPRs = 512 VGPRs, register-resident all scan.
    v16bf bw[4][16];
    #pragma unroll
    for (int g = 0; g < 4; ++g)
        #pragma unroll
        for (int kt = 0; kt < 16; ++kt) {
            const unsigned short* wp =
                pw + (size_t)(((g * 32 + s * 4 + w) * 16 + kt) * 512) + lane * 16;
            bw[g][kt] = __builtin_bit_cast(v16bf, *(const u32x8*)wp);
        }

    // ---- per-lane x-weights / biases (column = cb + laneN), time-invariant
    float wxv[4], bbv[4];
    {
        int col = cb + laneN;
        wxv[0] = Wxg[col]; wxv[1] = Wxi[col]; wxv[2] = Wxf[col]; wxv[3] = Wxo[col];
        bbv[0] = bg[col];  bbv[1] = bi[col];  bbv[2] = bf_[col]; bbv[3] = bo[col];
    }

    float cc[8], hn[8];
    #pragma unroll
    for (int r = 0; r < 8; ++r) { cc[r] = 0.0f; hn[r] = 0.0f; }

    unsigned short* hb0 = hb;
    unsigned short* hb1 = hb + (size_t)Bsz * H;

    #pragma unroll 1
    for (int t = 0; t < T; ++t) {
        const unsigned short* hin  = (t & 1) ? hb1 : hb0;
        unsigned short*       hout = (t & 1) ? hb0 : hb1;

        // stage full h tile for this batch group (16 KB) + x column
        {
            const u32x4* src = (const u32x4*)(hin + (size_t)b0 * H);
            for (int v = tid; v < TB * H / 8; v += NTHR)
                ((u32x4*)hsh)[v] = src[v];
        }
        if (tid < TB) xsh[tid] = x[(b0 + tid) * T + t];
        __syncthreads();

        // accumulator init: rank-1 x contribution + bias
        float xr[8];
        #pragma unroll
        for (int r = 0; r < 8; ++r) xr[r] = xsh[r + 8 * laneHi];
        v8f acc[4];
        #pragma unroll
        for (int g = 0; g < 4; ++g)
            #pragma unroll
            for (int r = 0; r < 8; ++r)
                acc[g][r] = xr[r] * wxv[g] + bbv[g];

        // K loop: A fragment from LDS, B from registers, 4 indep acc chains
        #pragma unroll
        for (int kt = 0; kt < 16; ++kt) {
            const unsigned short* hp = &hsh[laneN * H + kt * 32 + 16 * laneHi];
            A32 araw;
            araw.lo = *(const u32x4*)(hp);
            araw.hi = *(const u32x4*)(hp + 8);
            v16bf afrag = __builtin_bit_cast(v16bf, araw);
            #pragma unroll
            for (int g = 0; g < 4; ++g)
                acc[g] = __builtin_amdgcn_wmma_f32_16x16x32_bf16(
                    false, afrag, false, bw[g][kt], (short)0, acc[g],
                    false, false);
        }

        // elementwise gates; c persists in VGPRs
        #pragma unroll
        for (int r = 0; r < 8; ++r) {
            float gg = ftanh(acc[0][r]);
            float ii = sigm(acc[1][r]);
            float ff = sigm(acc[2][r]);
            float oo = sigm(acc[3][r]);
            float cn = gg * ii + cc[r] * ff;
            cc[r] = cn;
            hn[r] = ftanh(cn) * oo;
        }

        // publish this slice's new h (disjoint columns per wave/WG)
        #pragma unroll
        for (int r = 0; r < 8; ++r)
            hout[(size_t)(b0 + r + 8 * laneHi) * H + cb + laneN] = f2bf(hn[r]);

        // inter-WG barrier for this batch group (monotonic counter)
        __threadfence();
        __syncthreads();
        if (tid == 0) {
            __hip_atomic_fetch_add(mycnt, 1u, __ATOMIC_RELEASE,
                                   __HIP_MEMORY_SCOPE_AGENT);
            unsigned int target = (unsigned int)(NS * (t + 1));
            while (__hip_atomic_load(mycnt, __ATOMIC_ACQUIRE,
                                     __HIP_MEMORY_SCOPE_AGENT) < target)
                __builtin_amdgcn_s_sleep(2);
        }
        __syncthreads();
        __threadfence();
    }
}

// ---------------------------------------------------------------------------
// out = h_T @ W_ph + b_p   (256x512)(512x10); h_T is bf16 in hb[0] (T even).
// ---------------------------------------------------------------------------
__global__ void lstm_proj(const unsigned short* __restrict__ hT,
                          const float* __restrict__ Wp,
                          const float* __restrict__ bp,
                          float* __restrict__ out) {
    int idx = blockIdx.x * blockDim.x + threadIdx.x;
    if (idx >= Bsz * Cout) return;
    int b = idx / Cout, c = idx % Cout;
    float acc = bp[c];
    #pragma unroll 8
    for (int k = 0; k < H; ++k) acc += bf2f(hT[b * H + k]) * Wp[k * Cout + c];
    out[idx] = acc;
}

extern "C" void kernel_launch(void* const* d_in, const int* in_sizes, int n_in,
                              void* d_out, int out_size, void* d_ws, size_t ws_size,
                              hipStream_t stream) {
    const float* x   = (const float*)d_in[0];
    const float* Wgx = (const float*)d_in[1];
    const float* Wgh = (const float*)d_in[2];
    const float* Wix = (const float*)d_in[3];
    const float* Wih = (const float*)d_in[4];
    const float* Wfx = (const float*)d_in[5];
    const float* Wfh = (const float*)d_in[6];
    const float* Wox = (const float*)d_in[7];
    const float* Woh = (const float*)d_in[8];
    const float* Wph = (const float*)d_in[9];
    const float* bg  = (const float*)d_in[10];
    const float* bi  = (const float*)d_in[11];
    const float* bff = (const float*)d_in[12];
    const float* bo  = (const float*)d_in[13];
    const float* bp  = (const float*)d_in[14];

    // workspace layout
    unsigned short* pw  = (unsigned short*)d_ws;                        // 2 MB
    unsigned short* hb  = (unsigned short*)((char*)d_ws + (size_t)4 * H * H * 2);
    unsigned int*   cnt = (unsigned int*)((char*)hb + (size_t)2 * Bsz * H * 2);
    float* out = (float*)d_out;

    // zero h double-buffer + barrier counters (contiguous in ws)
    int zwords = (2 * Bsz * H * 2 + 16 * 4) / 4;
    lstm_init<<<(zwords + 255) / 256, 256, 0, stream>>>((unsigned int*)hb, zwords);

    lstm_pack_weights<<<(4 * H * H + 255) / 256, 256, 0, stream>>>(
        Wgh, Wih, Wfh, Woh, pw);

    lstm_scan_split<<<(Bsz / TB) * NS, NTHR, 0, stream>>>(
        x, Wgx, Wix, Wfx, Wox, bg, bi, bff, bo, pw, hb, cnt);

    lstm_proj<<<(Bsz * Cout + 255) / 256, 256, 0, stream>>>(hb, Wph, bp, out);
}
